// TransformerV5_53060025975280
// MI455X (gfx1250) — matrix operand order repaired and checked by reading, not compiled, not run on hardware
//
#include <hip/hip_runtime.h>

// ---------------------------------------------------------------------------
// MI455X (gfx1250) implementation of TransformerV5 texture-transfer block.
// wave32, WMMA bf16 16x16x32 with f32 accumulation everywhere.
// The 4x4096x4096x576 correlation GEMM is fused with streaming top-5 so the
// 256MB R matrix is never materialized (register top-k + shfl_xor merge).
// ---------------------------------------------------------------------------

typedef __bf16 bf16;
typedef __attribute__((ext_vector_type(16))) __bf16 v16bf;
typedef __attribute__((ext_vector_type(8)))  __bf16 v8bf;
typedef __attribute__((ext_vector_type(8)))  float  v8f;

namespace {
constexpr int kB = 4, kC = 64;
constexpr int kL = 4096;           // H*W
constexpr int kN = kB * kL;        // 16384 pixels total
constexpr int kKunf = kC * 9;      // 576
}

__device__ __forceinline__ int refl64(int i) {
  if (i < 0) i = -i;
  if (i >= 64) i = 126 - i;
  return i;
}

// ---- WMMA fragment loaders (contiguous k = 16*hs + e order, bf16, K-contiguous rows) ---
// A (16x32): lane lr holds row M=lr; hs=lane>=16 selects K halves
//   elements e0..e7  = K[k0+16*hs   .. +7]
//   elements e8..e15 = K[k0+16*hs+8 .. +7]
__device__ __forceinline__ v16bf load_a_frag(const bf16* __restrict__ row,
                                             int k0, int hs) {
  v8bf lo = *(const v8bf*)(row + k0 + 16 * hs);
  v8bf hi = *(const v8bf*)(row + k0 + 16 * hs + 8);
  v16bf r;
#pragma unroll
  for (int i = 0; i < 8; ++i) { r[i] = lo[i]; r[i + 8] = hi[i]; }
  return r;
}
// B (32x16): lane lr holds column N=lr; one contiguous 32B chunk of K
__device__ __forceinline__ v16bf load_b_frag(const bf16* __restrict__ row,
                                             int k0, int hs) {
  return *(const v16bf*)(row + k0 + 16 * hs);
}

// Sorted (descending) top-5 insertion with only static indexing -> stays in
// VGPRs, no scratch.
__device__ __forceinline__ void topk_insert(float (&v)[5], int (&x)[5],
                                            float nv, int ni) {
  if (nv > v[4]) {
    v[4] = nv; x[4] = ni;
#pragma unroll
    for (int j = 4; j > 0; --j) {
      if (v[j] > v[j - 1]) {
        float tf = v[j]; v[j] = v[j - 1]; v[j - 1] = tf;
        int   tt = x[j]; x[j] = x[j - 1]; x[j - 1] = tt;
      }
    }
  }
}

// ------------------------------- small kernels ------------------------------
__global__ void cvt_bf16_kernel(const float* __restrict__ s,
                                bf16* __restrict__ d, int n) {
  int t = blockIdx.x * blockDim.x + threadIdx.x;
  if (t < n) d[t] = (bf16)s[t];
}

// im2col with ZERO pad (conv path), output rows = pixel, cols = c*9+kh*3+kw
__global__ void im2col_kernel(const float* __restrict__ src,
                              bf16* __restrict__ dst) {
  size_t t = (size_t)blockIdx.x * blockDim.x + threadIdx.x;
  if (t >= (size_t)kN * kKunf) return;
  int k = (int)(t % kKunf);
  int n = (int)(t / kKunf);
  int b = n >> 12, p = n & 4095;
  int y = p >> 6, x = p & 63;
  int c = k / 9, r = k % 9, kh = r / 3, kw = r % 3;
  int iy = y + kh - 1, ix = x + kw - 1;
  float v = 0.f;
  if ((unsigned)iy < 64u && (unsigned)ix < 64u)
    v = src[(((size_t)b * kC + c) << 12) + iy * 64 + ix];
  dst[t] = (bf16)v;
}

// Normalized reflect-padded patch matrix qn[b][l][576] (bf16, K-contiguous)
__global__ void qn_build_kernel(const float* __restrict__ x,
                                bf16* __restrict__ qn) {
  int t = blockIdx.x * blockDim.x + threadIdx.x;
  if (t >= kB * kL) return;
  int b = t >> 12, l = t & 4095;
  int y = l >> 6, xx = l & 63;
  const float* xb = x + ((size_t)b * kC << 12);
  float ss = 0.f;
  for (int c = 0; c < kC; ++c) {
    const float* xc = xb + ((size_t)c << 12);
    for (int kh = 0; kh < 3; ++kh)
      for (int kw = 0; kw < 3; ++kw) {
        float v = xc[refl64(y + kh - 1) * 64 + refl64(xx + kw - 1)];
        ss += v * v;
      }
  }
  float inv = 1.f / fmaxf(sqrtf(ss), 1e-12f);
  bf16* q = qn + (size_t)t * kKunf;
  int k = 0;
  for (int c = 0; c < kC; ++c) {
    const float* xc = xb + ((size_t)c << 12);
    for (int kh = 0; kh < 3; ++kh)
      for (int kw = 0; kw < 3; ++kw)
        q[k++] = (bf16)(xc[refl64(y + kh - 1) * 64 + refl64(xx + kw - 1)] * inv);
  }
}

// x into the middle 64 channels of the channels-last concat buffer (ld=192)
__global__ void pack_x_kernel(const float* __restrict__ x,
                              bf16* __restrict__ catB) {
  int t = blockIdx.x * blockDim.x + threadIdx.x;
  if (t >= kB * kC * kL) return;
  int b = t >> 18, c = (t >> 12) & 63, p = t & 4095;
  catB[(size_t)(b * kL + p) * 192 + 64 + c] = (bf16)x[t];
}

// -------------------------- generic WMMA GEMM -------------------------------
// One wave per 16x16 output tile. A: [M][K] bf16 row-major (weights),
// B: [N][K] bf16 row-major (activations, K-contiguous). M = 16*mtiles = 64.
// mode 0: f32 NCHW output (+ optional relu). mode 1: bf16 channels-last at
// column offset coff with row stride ldc.
__global__ void gemm_bf16_wmma(const bf16* __restrict__ A, int lda,
                               const bf16* __restrict__ Bm, int ldb, int K,
                               const float* __restrict__ bias,
                               int relu, int mode,
                               float* __restrict__ outF,
                               bf16* __restrict__ outH, int ldc, int coff,
                               int mtiles, int ntiles) {
  int gid = blockIdx.x * blockDim.x + threadIdx.x;
  int wave = gid >> 5;
  if (wave >= mtiles * ntiles) return;  // wave-uniform: EXEC all-1s for WMMA
  int mt = wave % mtiles;
  int nt = wave / mtiles;
  int lane = threadIdx.x & 31;
  int hs = lane >> 4, lr = lane & 15;
  const bf16* arow = A  + (size_t)(mt * 16 + lr) * lda;
  const bf16* brow = Bm + (size_t)(nt * 16 + lr) * ldb;
  v8f acc = {};
  for (int k0 = 0; k0 < K; k0 += 32) {
    v16bf a = load_a_frag(arow, k0, hs);
    v16bf b = load_b_frag(brow, k0, hs);
    acc = __builtin_amdgcn_wmma_f32_16x16x32_bf16(false, a, false, b,
                                                  (short)0, acc, false, false);
  }
  int n = nt * 16 + lr;
#pragma unroll
  for (int v = 0; v < 8; ++v) {
    int o = mt * 16 + v + 8 * hs;           // output channel
    float r = acc[v] + bias[o];
    if (relu) r = fmaxf(r, 0.f);
    if (mode == 0) {
      int bb = n >> 12, p = n & 4095;       // NCHW f32
      outF[(((size_t)bb * kC + o) << 12) + p] = r;
    } else {
      outH[(size_t)n * ldc + coff + o] = (bf16)r;
    }
  }
}

// ------------------ fused correlation GEMM + streaming top-5 ----------------
// One wave per 16-row m-tile (1024 waves total). A-frags (18 x v16bf) are
// preloaded once (reused over all 256 n-tiles). Per n-tile: 18 WMMAs then a
// register top-5 insert per accumulator row; final cross-lane merge via
// shfl_xor butterfly inside each 16-lane half.
__global__ void corr_topk_kernel(const bf16* __restrict__ qn,
                                 float* __restrict__ topv,
                                 int* __restrict__ topi) {
  int wid = threadIdx.x >> 5;
  int gw = blockIdx.x * (blockDim.x >> 5) + wid;   // 0..1023
  int bb = gw >> 8;
  int mt = gw & 255;
  int lane = threadIdx.x & 31;
  int hs = lane >> 4, lr = lane & 15;
  const bf16* Q = qn + (size_t)bb * kL * kKunf;
  const bf16* arow = Q + (size_t)(mt * 16 + lr) * kKunf;

  v16bf afrag[18];
#pragma unroll
  for (int k = 0; k < 18; ++k) afrag[k] = load_a_frag(arow, k * 32, hs);

  float tv[8][5]; int ti[8][5];
#pragma unroll
  for (int v = 0; v < 8; ++v)
#pragma unroll
    for (int j = 0; j < 5; ++j) { tv[v][j] = -3.0e38f; ti[v][j] = 0; }

  for (int nt = 0; nt < 256; ++nt) {
    const bf16* brow = Q + (size_t)(nt * 16 + lr) * kKunf;
    if (nt + 1 < 256)  // speculative prefetch of next B row (global_prefetch)
      __builtin_prefetch(Q + (size_t)((nt + 1) * 16 + lr) * kKunf, 0, 0);
    v8f acc = {};
#pragma unroll
    for (int k = 0; k < 18; ++k) {
      v16bf bfrag = load_b_frag(brow, k * 32, hs);
      acc = __builtin_amdgcn_wmma_f32_16x16x32_bf16(false, afrag[k], false,
                                                    bfrag, (short)0, acc,
                                                    false, false);
    }
    int ncol = nt * 16 + lr;
#pragma unroll
    for (int v = 0; v < 8; ++v) topk_insert(tv[v], ti[v], acc[v], ncol);
  }

  // Butterfly merge: row M=v+8*hs is striped over the 16 lanes of one half.
#pragma unroll
  for (int v = 0; v < 8; ++v) {
#pragma unroll
    for (int mask = 1; mask < 16; mask <<= 1) {
      float ov[5]; int oi[5];
#pragma unroll
      for (int j = 0; j < 5; ++j) {
        ov[j] = __shfl_xor(tv[v][j], mask, 32);
        oi[j] = __shfl_xor(ti[v][j], mask, 32);
      }
#pragma unroll
      for (int j = 0; j < 5; ++j) topk_insert(tv[v], ti[v], ov[j], oi[j]);
    }
  }

  if (lr == 0) {  // lane 0 -> rows 0..7, lane 16 -> rows 8..15
#pragma unroll
    for (int v = 0; v < 8; ++v) {
      int m = mt * 16 + v + 8 * hs;
#pragma unroll
      for (int j = 0; j < 5; ++j) {
        topv[((size_t)bb * 5 + j) * kL + m] = tv[v][j];
        topi[((size_t)bb * 5 + j) * kL + m] = ti[v][j];
      }
    }
  }
}

// ---------------- gather + fold3x3 + score scale (ranks 1..4) ---------------
// Ts channels-last bf16 [b][p][(i-1)*64 + c], row stride 256.
__global__ void build_ts_kernel(const float* __restrict__ x,
                                const float* __restrict__ topv,
                                const int* __restrict__ topi,
                                bf16* __restrict__ tsB) {
  int t = blockIdx.x * blockDim.x + threadIdx.x;
  if (t >= kB * 4 * kC * kL) return;
  int p = t & 4095;
  int c = (t >> 12) & 63;
  int i = (t >> 18) & 3;     // rank-1 index: rank = i+1
  int b = t >> 20;
  int y = p >> 6, xx = p & 63;
  const int*   idx = topi + ((size_t)b * 5 + (i + 1)) * kL;
  const float* sc  = topv + ((size_t)b * 5 + (i + 1)) * kL;
  const float* xc  = x + (((size_t)b * kC + c) << 12);
  float acc = 0.f;
  for (int kh = 0; kh < 3; ++kh) {
    int yy = y + 1 - kh;
    if ((unsigned)yy >= 64u) continue;
    for (int kw = 0; kw < 3; ++kw) {
      int xx2 = xx + 1 - kw;
      if ((unsigned)xx2 >= 64u) continue;
      int m = idx[yy * 64 + xx2];
      int my = m >> 6, mx = m & 63;
      acc += xc[refl64(my + kh - 1) * 64 + refl64(mx + kw - 1)];
    }
  }
  float r = (acc * (1.f / 9.f)) * sc[p];
  tsB[(size_t)(b * kL + p) * 256 + i * 64 + c] = (bf16)r;
}

// ------------------------------- launcher -----------------------------------
extern "C" void kernel_launch(void* const* d_in, const int* in_sizes, int n_in,
                              void* d_out, int out_size, void* d_ws,
                              size_t ws_size, hipStream_t stream) {
  (void)in_sizes; (void)n_in; (void)out_size; (void)ws_size;
  const float* x   = (const float*)d_in[0];
  const float* w1  = (const float*)d_in[1];
  const float* b1  = (const float*)d_in[2];
  const float* w2  = (const float*)d_in[3];
  const float* b2  = (const float*)d_in[4];
  const float* wt1 = (const float*)d_in[5];
  const float* bt1 = (const float*)d_in[6];
  const float* wt2 = (const float*)d_in[7];
  const float* bt2 = (const float*)d_in[8];
  float* out = (float*)d_out;

  char* ws = (char*)d_ws;
  size_t off = 0;
  auto wsalloc = [&](size_t bytes) -> void* {
    void* p = (void*)(ws + off);
    off += (bytes + 255) & ~(size_t)255;
    return p;
  };
  // ~58 MB total workspace
  bf16*  qn   = (bf16*)wsalloc((size_t)kB * kL * kKunf * 2);  // 18.9 MB
  bf16*  col  = (bf16*)wsalloc((size_t)kN * kKunf * 2);       // 18.9 MB
  float* f1   = (float*)wsalloc((size_t)kB * kC * kL * 4);    //  4.2 MB
  bf16*  catB = (bf16*)wsalloc((size_t)kN * 192 * 2);         //  6.3 MB
  bf16*  tsB  = (bf16*)wsalloc((size_t)kN * 256 * 2);         //  8.4 MB
  float* topv = (float*)wsalloc((size_t)kB * 5 * kL * 4);
  int*   topi = (int*)wsalloc((size_t)kB * 5 * kL * 4);
  bf16*  w1b  = (bf16*)wsalloc((size_t)64 * 576 * 2);
  bf16*  w2b  = (bf16*)wsalloc((size_t)64 * 576 * 2);
  bf16*  wt1b = (bf16*)wsalloc((size_t)64 * 256 * 2);
  bf16*  wt2b = (bf16*)wsalloc((size_t)64 * 192 * 2);

  // weights -> bf16 (K-contiguous row-major already matches OIHW flatten)
  cvt_bf16_kernel<<<144, 256, 0, stream>>>(w1, w1b, 64 * 576);
  cvt_bf16_kernel<<<144, 256, 0, stream>>>(w2, w2b, 64 * 576);
  cvt_bf16_kernel<<<64, 256, 0, stream>>>(wt1, wt1b, 64 * 256);
  cvt_bf16_kernel<<<48, 256, 0, stream>>>(wt2, wt2b, 64 * 192);

  // conv1 (3x3, zero pad) -> relu -> f1 (NCHW f32)
  im2col_kernel<<<36864, 256, 0, stream>>>(x, col);
  gemm_bf16_wmma<<<512, 256, 0, stream>>>(w1b, 576, col, 576, 576, b1,
                                          /*relu=*/1, /*mode=*/0, f1, nullptr,
                                          0, 0, 4, 1024);
  // conv2 -> feature into concat buffer channels [0,64)
  im2col_kernel<<<36864, 256, 0, stream>>>(f1, col);
  gemm_bf16_wmma<<<512, 256, 0, stream>>>(w2b, 576, col, 576, 576, b2,
                                          0, 1, nullptr, catB, 192, 0, 4, 1024);
  // x into concat channels [64,128)
  pack_x_kernel<<<4096, 256, 0, stream>>>(x, catB);

  // correlation path: qn, fused GEMM+top5, gather/fold/scale
  qn_build_kernel<<<64, 256, 0, stream>>>(x, qn);
  corr_topk_kernel<<<128, 256, 0, stream>>>(qn, topv, topi);
  build_ts_kernel<<<16384, 256, 0, stream>>>(x, topv, topi, tsB);

  // texture 1x1 conv (K=256) -> concat channels [128,192)
  gemm_bf16_wmma<<<512, 256, 0, stream>>>(wt1b, 256, tsB, 256, 256, bt1,
                                          0, 1, nullptr, catB, 192, 128, 4, 1024);
  // final 1x1 conv (K=192) -> y (NCHW f32)
  gemm_bf16_wmma<<<512, 256, 0, stream>>>(wt2b, 192, catB, 192, 192, bt2,
                                          0, 0, out, nullptr, 0, 0, 4, 1024);
}
